// FluxSingleStreamBlock_53884659696162
// MI455X (gfx1250) — matrix operand order repaired and checked by reading, not compiled, not run on hardware
//
#include <hip/hip_runtime.h>
#include <math.h>

// ---------------- problem constants ----------------
#define B_SZ 1
#define L_SEQ 2048
#define HID 3072
#define NH 24
#define HD 128
#define MLP_DIM 12288
#define QKV_DIM (3*HID)

typedef __attribute__((ext_vector_type(16))) __bf16 v16bf;
typedef __attribute__((ext_vector_type(8)))  float  v8f;
typedef __attribute__((ext_vector_type(4)))  unsigned u32x4;
typedef __attribute__((ext_vector_type(8)))  int      i32x8;
typedef __attribute__((ext_vector_type(4)))  int      i32x4;

#if __has_builtin(__builtin_amdgcn_tensor_load_to_lds)
#define HAVE_TDM 1
#else
#define HAVE_TDM 0
#endif

#if __has_include(<hip/amd_detail/amd_gfx1250_TDM.h>)
#define TDM_6ARG 1
#else
#define TDM_6ARG 0
#endif

union AFrag { v16bf v; uint4 u[2]; };

__device__ __forceinline__ __bf16 f2bf(float x) {
    union { float f; unsigned u; } a; a.f = x;
    unsigned r = (a.u + 0x7FFFu + ((a.u >> 16) & 1u)) >> 16;
    union { unsigned short s; __bf16 b; } o; o.s = (unsigned short)r;
    return o.b;
}

__device__ __forceinline__ float halfmax16(float v) {
    v = fmaxf(v, __shfl_xor(v, 1, 32));
    v = fmaxf(v, __shfl_xor(v, 2, 32));
    v = fmaxf(v, __shfl_xor(v, 4, 32));
    v = fmaxf(v, __shfl_xor(v, 8, 32));
    return v;
}
__device__ __forceinline__ float halfsum16(float v) {
    v += __shfl_xor(v, 1, 32);
    v += __shfl_xor(v, 2, 32);
    v += __shfl_xor(v, 4, 32);
    v += __shfl_xor(v, 8, 32);
    return v;
}

__device__ __forceinline__ unsigned lds_off(const void* p) {
    return (unsigned)(unsigned long long)(uintptr_t)p;
}

#if HAVE_TDM
// Issue a TDM 2D tile load: BK x BM tile of bf16, row stride K elements.
// D# built per CDNA5 ISA 8.3/8.4 (group0: flags/lds/global/type, group1: dims).
__device__ __forceinline__ void tdm_load_tile(const __bf16* gptr, unsigned ldsoff,
                                              int K, int tile_k, int tile_m) {
    unsigned long long ga = (unsigned long long)(uintptr_t)gptr;
    u32x4 g0;
    g0[0] = 1u;                                   // count=1, user descriptor
    g0[1] = ldsoff;                               // lds_addr (bytes)
    g0[2] = (unsigned)ga;                         // global_addr[31:0]
    g0[3] = (unsigned)(ga >> 32) | 0x80000000u;   // global_addr[56:32] | type=2
    i32x8 g1;
    g1[0] = (int)(1u << 16);                      // wg_mask=0, data_size=1 (2 bytes)
    g1[1] = (int)(((unsigned)K & 0xFFFFu) << 16); // tensor_dim0[15:0]
    g1[2] = (int)((((unsigned)K >> 16) & 0xFFFFu) |
                  (((unsigned)tile_m & 0xFFFFu) << 16)); // tdim0[31:16] | tensor_dim1[15:0]
    g1[3] = (int)(((unsigned)tile_k & 0xFFFFu) << 16);   // tensor_dim1[31:16]=0 | tile_dim0
    g1[4] = (int)((unsigned)tile_m & 0xFFFFu);           // tile_dim1 | tile_dim2=0
    g1[5] = (int)(unsigned)K;                             // tensor_dim0_stride[31:0]
    g1[6] = 0;                                            // stride0[47:32] | stride1[15:0]
    g1[7] = 0;
    i32x4 z4 = {0, 0, 0, 0};
#if TDM_6ARG
    i32x8 z8 = {0, 0, 0, 0, 0, 0, 0, 0};
    __builtin_amdgcn_tensor_load_to_lds(g0, g1, z4, z4, z8, 0);
#else
    __builtin_amdgcn_tensor_load_to_lds(g0, g1, z4, z4, 0);
#endif
}
__device__ __forceinline__ void tdm_wait0() {
#if __has_builtin(__builtin_amdgcn_s_wait_tensorcnt)
    __builtin_amdgcn_s_wait_tensorcnt(0);
#else
    asm volatile("s_wait_tensorcnt 0x0" ::: "memory");
#endif
}
#endif

// ---------------- kernel 1: mod = silu(vec) @ w_mod + b_mod ----------------
__global__ __launch_bounds__(256)
void mod_gemv(const float* __restrict__ vec, const float* __restrict__ w,
              const float* __restrict__ b, float* __restrict__ mod) {
    __shared__ float sv[HID];
    int tid = threadIdx.x;
    for (int i = tid; i < HID; i += 256) {
        float x = vec[i];
        sv[i] = x / (1.0f + __expf(-x));
    }
    __syncthreads();
    int j = blockIdx.x * 256 + tid;
    float acc = 0.0f;
    for (int i = 0; i < HID; ++i)
        acc += sv[i] * w[(size_t)i * QKV_DIM + j];
    mod[j] = acc + b[j];
}

// ---------------- kernel 2: layernorm + modulation -> bf16 x_mod ----------------
__global__ __launch_bounds__(256)
void ln_mod(const float* __restrict__ x, const float* __restrict__ mod,
            __bf16* __restrict__ xmod) {
    int m = blockIdx.x, tid = threadIdx.x;
    __shared__ float red[8];
    const float* xr = x + (size_t)m * HID;
    float s = 0.0f;
    for (int i = tid; i < HID; i += 256) s += xr[i];
    for (int mm = 16; mm >= 1; mm >>= 1) s += __shfl_xor(s, mm, 32);
    if ((tid & 31) == 0) red[tid >> 5] = s;
    __syncthreads();
    float mu = 0.0f;
    #pragma unroll
    for (int i = 0; i < 8; ++i) mu += red[i];
    mu /= (float)HID;
    __syncthreads();
    float v = 0.0f;
    for (int i = tid; i < HID; i += 256) { float d = xr[i] - mu; v += d * d; }
    for (int mm = 16; mm >= 1; mm >>= 1) v += __shfl_xor(v, mm, 32);
    if ((tid & 31) == 0) red[tid >> 5] = v;
    __syncthreads();
    float var = 0.0f;
    #pragma unroll
    for (int i = 0; i < 8; ++i) var += red[i];
    var /= (float)HID;
    float inv = rsqrtf(var + 1e-6f);
    __bf16* xo = xmod + (size_t)m * HID;
    for (int i = tid; i < HID; i += 256) {
        float val = (xr[i] - mu) * inv * (1.0f + mod[HID + i]) + mod[i];
        xo[i] = f2bf(val);
    }
}

// ---------------- WMMA GEMM: C = A(bf16, MxK) @ Bw(f32->bf16, KxN) + bias ----------------
// EPI 0: fp32 out   EPI 1: gelu -> bf16 out   EPI 2: out = x + gate*(C+bias+mlp)
#define BM 128
#define BN 128
#define BK 32

template<int EPI>
__global__ __launch_bounds__(256)
void gemm_wmma(const __bf16* __restrict__ A, const float* __restrict__ Bw,
               const float* __restrict__ bias,
               float* __restrict__ Cf, __bf16* __restrict__ Cb,
               const float* __restrict__ mlp, const float* __restrict__ xres,
               const float* __restrict__ gate,
               int M, int N, int K) {
    __shared__ __bf16 As[2][BM * BK];   // [m][k], TDM double buffer
    __shared__ __bf16 Bs[BN * BK];      // [n][k] (transposed)
    const int tid  = threadIdx.x;
    const int lane = tid & 31;
    const int wid  = tid >> 5;
    const int wrow = wid >> 2;       // 0..1 -> 64-row slab
    const int wcol = wid & 3;        // 0..3 -> 32-col slab
    const int m0 = blockIdx.y * BM;
    const int n0 = blockIdx.x * BN;
    const int hlo = (lane < 16) ? 0 : 1;
    const int l15 = lane & 15;

    v8f acc[4][2];
    #pragma unroll
    for (int i = 0; i < 4; ++i)
        #pragma unroll
        for (int j = 0; j < 2; ++j) acc[i][j] = (v8f)(0.0f);

    const int nk = K / BK;
    const int arow = tid >> 1;
    const int acol = (tid & 1) * 16;

#if HAVE_TDM
    // prologue: async DMA of first A tile via Tensor Data Mover
    if (wid == 0)
        tdm_load_tile(A + (size_t)m0 * K, lds_off(&As[0][0]), K, BK, BM);
#endif

    for (int kt = 0; kt < nk; ++kt) {
        const int cur = kt & 1;
        // stage B tile: fp32 -> bf16, transpose to [n][k]
        #pragma unroll
        for (int it = 0; it < 4; ++it) {
            int idx = it * 1024 + tid * 4;
            int br = idx >> 7;       // k
            int bc = idx & 127;      // n
            const float4 f = *reinterpret_cast<const float4*>(
                Bw + (size_t)(kt * BK + br) * N + n0 + bc);
            Bs[(bc + 0) * BK + br] = f2bf(f.x);
            Bs[(bc + 1) * BK + br] = f2bf(f.y);
            Bs[(bc + 2) * BK + br] = f2bf(f.z);
            Bs[(bc + 3) * BK + br] = f2bf(f.w);
        }
#if HAVE_TDM
        if (wid == 0) tdm_wait0();          // A[cur] DMA complete
        __syncthreads();                    // A + B visible to all waves
        if (wid == 0 && kt + 1 < nk)        // overlap next tile DMA with compute
            tdm_load_tile(A + (size_t)m0 * K + (kt + 1) * BK,
                          lds_off(&As[cur ^ 1][0]), K, BK, BM);
#else
        // manual A stage: 16 bf16 per thread (2 x b128)
        {
            const uint4* src = reinterpret_cast<const uint4*>(
                A + (size_t)(m0 + arow) * K + kt * BK + acol);
            uint4* dst = reinterpret_cast<uint4*>(&As[cur][arow * BK + acol]);
            dst[0] = src[0];
            dst[1] = src[1];
        }
        __syncthreads();
#endif
        if (kt + 1 < nk)
            __builtin_prefetch(Bw + (size_t)((kt + 1) * BK + (tid >> 3)) * N + n0 + (tid & 7) * 16, 0, 1);

        AFrag af[4], bf[2];
        #pragma unroll
        for (int mi = 0; mi < 4; ++mi) {
            int row = wrow * 64 + mi * 16 + l15;
            const uint4* p = reinterpret_cast<const uint4*>(&As[cur][row * BK]);
            af[mi].u[0] = p[hlo];       // K 0-7 / 8-15
            af[mi].u[1] = p[2 + hlo];   // K 16-23 / 24-31
        }
        #pragma unroll
        for (int ni = 0; ni < 2; ++ni) {
            int col = wcol * 32 + ni * 16 + l15;
            const uint4* p = reinterpret_cast<const uint4*>(&Bs[col * BK]);
            bf[ni].u[0] = p[hlo];       // K 0-7 / 8-15
            bf[ni].u[1] = p[2 + hlo];   // K 16-23 / 24-31
        }
        #pragma unroll
        for (int mi = 0; mi < 4; ++mi)
            #pragma unroll
            for (int ni = 0; ni < 2; ++ni)
                acc[mi][ni] = __builtin_amdgcn_wmma_f32_16x16x32_bf16(
                    false, af[mi].v, false, bf[ni].v,
                    (short)0, acc[mi][ni], false, false);
        __syncthreads();
    }

    // epilogue
    #pragma unroll
    for (int mi = 0; mi < 4; ++mi) {
        #pragma unroll
        for (int ni = 0; ni < 2; ++ni) {
            int colg = n0 + wcol * 32 + ni * 16 + l15;
            float bv = bias[colg];
            #pragma unroll
            for (int v = 0; v < 8; ++v) {
                int rowg = m0 + wrow * 64 + mi * 16 + v + hlo * 8;
                float val = acc[mi][ni][v] + bv;
                size_t o = (size_t)rowg * N + colg;
                if (EPI == 0) {
                    Cf[o] = val;
                } else if (EPI == 1) {
                    float g = 0.5f * val *
                        (1.0f + tanhf(0.7978845608f * (val + 0.044715f * val * val * val)));
                    Cb[o] = f2bf(g);
                } else {
                    float t = val + mlp[o];
                    Cf[o] = xres[o] + gate[colg] * t;
                }
            }
        }
    }
}

// ---------------- kernel 4: rmsnorm + rope + head-major bf16 pack ----------------
__device__ __forceinline__ float block_sum128(float v, float* red) {
    for (int m = 16; m >= 1; m >>= 1) v += __shfl_xor(v, m, 32);
    int tid = threadIdx.x;
    __syncthreads();
    if ((tid & 31) == 0) red[tid >> 5] = v;
    __syncthreads();
    return red[0] + red[1] + red[2] + red[3];
}

__global__ __launch_bounds__(128)
void qk_prep(const float* __restrict__ qkv, const float* __restrict__ pe,
             const float* __restrict__ nqw, const float* __restrict__ nkw,
             __bf16* __restrict__ qh, __bf16* __restrict__ kh, __bf16* __restrict__ vh) {
    int m = blockIdx.x, h = blockIdx.y, d = threadIdx.x;
    __shared__ float red[4];
    size_t base = (size_t)m * QKV_DIM + h * HD + d;
    float qv = qkv[base];
    float kv = qkv[base + HID];
    float vv = qkv[base + 2 * HID];
    float sq = block_sum128(qv * qv, red);
    float sk = block_sum128(kv * kv, red);
    const float eps = 1.1920929e-7f;
    qv = qv * rsqrtf(sq / (float)HD + eps) * nqw[d];
    kv = kv * rsqrtf(sk / (float)HD + eps) * nkw[d];
    // rope: pairs (2t, 2t+1); pe shape (L, HD/2, 2, 2)
    int t = d >> 1, r = d & 1;
    size_t pb = (((size_t)m * (HD / 2) + t) * 2 + r) * 2;
    float c0 = pe[pb + 0];
    float c1 = pe[pb + 1];
    float qp = __shfl_xor(qv, 1, 32);
    float kp = __shfl_xor(kv, 1, 32);
    float qe = (r == 0) ? qv : qp, qo = (r == 0) ? qp : qv;
    float ke = (r == 0) ? kv : kp, ko = (r == 0) ? kp : kv;
    float qn = c0 * qe + c1 * qo;
    float kn = c0 * ke + c1 * ko;
    size_t ob = ((size_t)h * L_SEQ + m) * HD + d;
    qh[ob] = f2bf(qn);
    kh[ob] = f2bf(kn);
    vh[ob] = f2bf(vv);
}

// ---------------- kernel 5: WMMA flash attention ----------------
// grid (L/64, NH), 128 threads = 4 waves; each wave owns 16 query rows.
__global__ __launch_bounds__(128)
void attn_wmma(const __bf16* __restrict__ qh, const __bf16* __restrict__ kh,
               const __bf16* __restrict__ vh, __bf16* __restrict__ attn) {
    __shared__ __bf16 Vt[HD * 32];        // [d][kk]
    __shared__ __bf16 Ps[4 * 16 * 32];    // per-wave P staging [row][kk]
    const int tid  = threadIdx.x;
    const int lane = tid & 31;
    const int wid  = tid >> 5;
    const int hlo  = (lane < 16) ? 0 : 1;
    const int l15  = lane & 15;
    const int h    = blockIdx.y;
    const int q0   = blockIdx.x * 64 + wid * 16;
    const float sc = 0.08838834764831845f; // 1/sqrt(128)

    // Q fragments (16x128 -> 4 A-frags of 16x32)
    AFrag qf[4];
    {
        int qrow = q0 + l15;
        #pragma unroll
        for (int c = 0; c < 4; ++c) {
            const uint4* p = reinterpret_cast<const uint4*>(
                qh + ((size_t)h * L_SEQ + qrow) * HD + c * 32);
            qf[c].u[0] = p[hlo];
            qf[c].u[1] = p[2 + hlo];
        }
    }

    float mrow[8], lrow[8];
    v8f o[8];
    #pragma unroll
    for (int v = 0; v < 8; ++v) { mrow[v] = -3.0e38f; lrow[v] = 0.0f; o[v] = (v8f)(0.0f); }

    __bf16* ps = &Ps[wid * 16 * 32];

    for (int kt = 0; kt < L_SEQ / 32; ++kt) {
        // cooperative V tile stage: [key 32][d 128] -> Vt[d][kk]
        #pragma unroll
        for (int i = 0; i < 4; ++i) {
            int idx = i * 128 + tid;          // 0..511
            int kk  = idx >> 4;               // 0..31
            int qi  = idx & 15;               // uint4 index along d
            uint4 dv = *reinterpret_cast<const uint4*>(
                vh + ((size_t)h * L_SEQ + kt * 32 + kk) * HD + qi * 8);
            const __bf16* e = reinterpret_cast<const __bf16*>(&dv);
            #pragma unroll
            for (int t2 = 0; t2 < 8; ++t2) Vt[(qi * 8 + t2) * 32 + kk] = e[t2];
        }
        __syncthreads();

        // S = Q @ K^T for 32 keys; interleave the two 16-key accumulators so
        // consecutive wmma target different D registers (hides RAW-on-C hazard)
        v8f s[2];
        s[0] = (v8f)(0.0f); s[1] = (v8f)(0.0f);
        #pragma unroll
        for (int c = 0; c < 4; ++c) {
            #pragma unroll
            for (int half = 0; half < 2; ++half) {
                AFrag kf;
                int key = kt * 32 + half * 16 + l15;
                const uint4* p = reinterpret_cast<const uint4*>(
                    kh + ((size_t)h * L_SEQ + key) * HD + c * 32);
                kf.u[0] = p[hlo];
                kf.u[1] = p[2 + hlo];
                s[half] = __builtin_amdgcn_wmma_f32_16x16x32_bf16(
                    false, qf[c].v, false, kf.v, (short)0, s[half], false, false);
            }
        }

        // online softmax (rows live in 16-lane halves)
        float rscale[8];
        #pragma unroll
        for (int v = 0; v < 8; ++v) {
            float s0 = s[0][v] * sc;
            float s1 = s[1][v] * sc;
            float mx = halfmax16(fmaxf(s0, s1));
            float nm = fmaxf(mrow[v], mx);
            float p0 = __expf(s0 - nm);
            float p1 = __expf(s1 - nm);
            float rs = halfsum16(p0 + p1);
            float sf = __expf(mrow[v] - nm);
            lrow[v] = lrow[v] * sf + rs;
            mrow[v] = nm;
            rscale[v] = sf;
            s[0][v] = p0;
            s[1][v] = p1;
        }
        #pragma unroll
        for (int n = 0; n < 8; ++n)
            #pragma unroll
            for (int v = 0; v < 8; ++v) o[n][v] *= rscale[v];

        // permute P (C-layout) -> A-frag layout via per-wave LDS
        #pragma unroll
        for (int half = 0; half < 2; ++half)
            #pragma unroll
            for (int v = 0; v < 8; ++v) {
                int row = v + hlo * 8;
                int col = half * 16 + l15;
                ps[row * 32 + col] = f2bf(s[half][v]);
            }
        asm volatile("s_wait_dscnt 0" ::: "memory");
        AFrag pf;
        {
            const uint4* p = reinterpret_cast<const uint4*>(&ps[l15 * 32]);
            pf.u[0] = p[hlo];
            pf.u[1] = p[2 + hlo];
        }

        // O += P @ V (8 d-chunks of 16, independent accumulators)
        #pragma unroll
        for (int n = 0; n < 8; ++n) {
            AFrag vf;
            int d = n * 16 + l15;
            const uint4* p = reinterpret_cast<const uint4*>(&Vt[d * 32]);
            vf.u[0] = p[hlo];
            vf.u[1] = p[2 + hlo];
            o[n] = __builtin_amdgcn_wmma_f32_16x16x32_bf16(
                false, pf.v, false, vf.v, (short)0, o[n], false, false);
        }
        __syncthreads();
    }

    // normalize + store bf16 attn[m][h*HD + d]
    #pragma unroll
    for (int n = 0; n < 8; ++n) {
        int d = n * 16 + l15;
        #pragma unroll
        for (int v = 0; v < 8; ++v) {
            int rowg = q0 + v + hlo * 8;
            float val = o[n][v] / lrow[v];
            attn[(size_t)rowg * HID + h * HD + d] = f2bf(val);
        }
    }
}

// ---------------- host-side orchestration ----------------
extern "C" void kernel_launch(void* const* d_in, const int* in_sizes, int n_in,
                              void* d_out, int out_size, void* d_ws, size_t ws_size,
                              hipStream_t stream) {
    const float* x      = (const float*)d_in[0];
    const float* vec    = (const float*)d_in[1];
    const float* pe     = (const float*)d_in[2];
    const float* w_mod  = (const float*)d_in[3];
    const float* b_mod  = (const float*)d_in[4];
    const float* w_qkv  = (const float*)d_in[5];
    const float* b_qkv  = (const float*)d_in[6];
    const float* w_fc1  = (const float*)d_in[7];
    const float* b_fc1  = (const float*)d_in[8];
    const float* w_fc2  = (const float*)d_in[9];
    const float* b_fc2  = (const float*)d_in[10];
    const float* w_out  = (const float*)d_in[11];
    const float* b_out  = (const float*)d_in[12];
    const float* nqw    = (const float*)d_in[13];
    const float* nkw    = (const float*)d_in[14];

    char* ws = (char*)d_ws;
    size_t off = 0;
    auto alloc = [&](size_t bytes) -> void* {
        void* p = ws + off;
        off = (off + bytes + 255) & ~(size_t)255;
        return p;
    };
    float*  mod   = (float*) alloc((size_t)QKV_DIM * 4);
    __bf16* xmod  = (__bf16*)alloc((size_t)L_SEQ * HID * 2);
    float*  qkv   = (float*) alloc((size_t)L_SEQ * QKV_DIM * 4);
    __bf16* qh    = (__bf16*)alloc((size_t)L_SEQ * HID * 2);
    __bf16* kh    = (__bf16*)alloc((size_t)L_SEQ * HID * 2);
    __bf16* vh    = (__bf16*)alloc((size_t)L_SEQ * HID * 2);
    __bf16* attnb = (__bf16*)alloc((size_t)L_SEQ * HID * 2);
    __bf16* h1    = (__bf16*)alloc((size_t)L_SEQ * MLP_DIM * 2);
    float*  mlp   = (float*) alloc((size_t)L_SEQ * HID * 4);
    float*  out   = (float*)d_out;
    const float* gate = mod + 2 * HID;

    // 1) modulation vector
    mod_gemv<<<QKV_DIM / 256, 256, 0, stream>>>(vec, w_mod, b_mod, mod);
    // 2) layernorm + scale/shift -> bf16
    ln_mod<<<L_SEQ, 256, 0, stream>>>(x, mod, xmod);
    // 3) qkv = xmod @ w_qkv + b_qkv  (fp32 out)
    gemm_wmma<0><<<dim3(QKV_DIM / BN, L_SEQ / BM), 256, 0, stream>>>(
        xmod, w_qkv, b_qkv, qkv, nullptr, nullptr, nullptr, nullptr,
        L_SEQ, QKV_DIM, HID);
    // 4) rmsnorm + rope + head-major bf16 pack
    qk_prep<<<dim3(L_SEQ, NH), HD, 0, stream>>>(qkv, pe, nqw, nkw, qh, kh, vh);
    // 5) flash attention
    attn_wmma<<<dim3(L_SEQ / 64, NH), 128, 0, stream>>>(qh, kh, vh, attnb);
    // 6) fc1 + gelu -> bf16
    gemm_wmma<1><<<dim3(MLP_DIM / BN, L_SEQ / BM), 256, 0, stream>>>(
        xmod, w_fc1, b_fc1, nullptr, h1, nullptr, nullptr, nullptr,
        L_SEQ, MLP_DIM, HID);
    // 7) fc2 -> fp32 mlp
    gemm_wmma<0><<<dim3(HID / BN, L_SEQ / BM), 256, 0, stream>>>(
        h1, w_fc2, b_fc2, mlp, nullptr, nullptr, nullptr, nullptr,
        L_SEQ, HID, MLP_DIM);
    // 8) out-proj + fused bias + mlp + gated residual -> d_out
    gemm_wmma<2><<<dim3(HID / BN, L_SEQ / BM), 256, 0, stream>>>(
        attnb, w_out, b_out, out, nullptr, mlp, x, gate,
        L_SEQ, HID, HID);
}
